// NERMOEE_11647951307121
// MI455X (gfx1250) — matrix operand order repaired and checked by reading, not compiled, not run on hardware
//
#include <hip/hip_runtime.h>
#include <hip/hip_bf16.h>

// ---------------------------------------------------------------------------
// MI455X (gfx1250) NER-MoE BiLSTM+attention pipeline, bf16 WMMA throughout,
// with gfx1250 async global->LDS staging (ASYNCcnt) when available.
// ---------------------------------------------------------------------------

typedef __attribute__((ext_vector_type(16))) __bf16 v16bf;
typedef __attribute__((ext_vector_type(8)))  float  v8f;

#define NER_B 32
#define NER_S 512
#define NER_E 768
#define NER_H 512
#define NER_D 288     // 9*32
#define NER_F 1152    // NHEAD*D
#define NER_G 2048    // 4*H

#if defined(__gfx1250__) && \
    __has_builtin(__builtin_amdgcn_global_load_async_to_lds_b128) && \
    __has_builtin(__builtin_amdgcn_s_wait_asynccnt)
#define HAVE_ASYNC_LDS 1
#else
#define HAVE_ASYNC_LDS 0
#endif

#if HAVE_ASYNC_LDS
typedef int async_v4i __attribute__((vector_size(16)));
typedef __attribute__((address_space(1))) async_v4i* g_v4i_ptr;
typedef __attribute__((address_space(3))) async_v4i* l_v4i_ptr;
__device__ __forceinline__ void async_b128(const void* g, void* l) {
  __builtin_amdgcn_global_load_async_to_lds_b128(
      (g_v4i_ptr)(unsigned long long)g,
      (l_v4i_ptr)(unsigned)(unsigned long long)l,
      0, 0);
}
#define WAIT_ASYNC(n) __builtin_amdgcn_s_wait_asynccnt(n)
#endif

__device__ __forceinline__ v8f wmma_bf16(v16bf a, v16bf b, v8f c) {
  return __builtin_amdgcn_wmma_f32_16x16x32_bf16(false, a, false, b, (short)0, c, false, false);
}

// 16-bit A-matrix 16x32 K index for element e of the v16bf (contiguous: 16*(lane/16) + e, matches B)
__device__ __forceinline__ int ka_off(int e, int lane) {
  return ((lane >> 4) << 4) + e;
}
// 16-bit B-matrix 32x16 K base for a lane
__device__ __forceinline__ int kb_off(int lane) { return (lane >> 4) << 4; }

__device__ __forceinline__ v16bf frag_a_lds(const __bf16* base, int stride,
                                            int m0, int k0, int lane) {
  const __bf16* p = base + (m0 + (lane & 15)) * stride + k0;
  v16bf a;
#pragma unroll
  for (int e = 0; e < 16; ++e) a[e] = p[ka_off(e, lane)];
  return a;
}
__device__ __forceinline__ v16bf frag_b_lds(const __bf16* base, int stride,
                                            int n0, int k0, int lane) {
  const __bf16* p = base + (n0 + (lane & 15)) * stride + k0 + kb_off(lane);
  v16bf b;
#pragma unroll
  for (int e = 0; e < 16; ++e) b[e] = p[e];
  return b;
}

// ---------------------------------------------------------------------------
// fp32 -> bf16 conversion / x transpose ([B,S,E] f32 -> [S,B,E] bf16)
// ---------------------------------------------------------------------------
__global__ void cvt_bf16_kernel(const float* __restrict__ in,
                                __bf16* __restrict__ out, long n) {
  long i = (long)blockIdx.x * blockDim.x + threadIdx.x;
  long stride = (long)gridDim.x * blockDim.x;
  for (; i < n; i += stride) out[i] = (__bf16)in[i];
}

__global__ void xpose_x_kernel(const float* __restrict__ x,
                               __bf16* __restrict__ xT) {
  int bid = blockIdx.x;            // b*S + s
  int b = bid >> 9, s = bid & 511;
  const float* src = x + ((long)b * NER_S + s) * NER_E;
  __bf16* dst = xT + ((long)s * NER_B + b) * NER_E;
  for (int i = threadIdx.x; i < NER_E; i += 256) dst[i] = (__bf16)src[i];
}

// ---------------------------------------------------------------------------
// Generic bf16 WMMA GEMM:  C[M,N] = A[M,K] * Bw[N,K]^T + bias[N]
// 128x64 tile / WG; 8 waves in 4x2 grid, each 32x32 (4 WMMA tiles);
// K in 32-chunks, double-buffered LDS with async global->LDS staging.
// Requires M%128==0, K%32==0; N arbitrary (bounds-checked at store; OOB B
// rows are clamped -- their product columns are discarded at the store).
// ---------------------------------------------------------------------------
#define G_BM 128
#define G_BN 64
#define G_LDT 48   // 32 + 16 pad, keeps rows 16B-aligned

__device__ __forceinline__ void stage_tile(const __bf16* __restrict__ A,
                                           const __bf16* __restrict__ Bw,
                                           __bf16 (*sA)[G_LDT],
                                           __bf16 (*sB)[G_LDT],
                                           int m_base, int n_base, int N, int K,
                                           int c, int tid) {
  const long kc = (long)c << 5;
#pragma unroll
  for (int k2 = 0; k2 < 2; ++k2) {           // A: 128x32 = 2 x b128 per thread
    int i = (tid + (k2 << 8)) << 3;
    int r = i >> 5, cl = i & 31;
    const __bf16* src = A + (long)(m_base + r) * K + kc + cl;
#if HAVE_ASYNC_LDS
    async_b128(src, &sA[r][cl]);
#else
    *(uint4*)&sA[r][cl] = *(const uint4*)src;
#endif
  }
  {                                          // B: 64x32 = 1 x b128 per thread
    int i = tid << 3;
    int r = i >> 5, cl = i & 31;
    int nrow = n_base + r;
    if (nrow >= N) nrow = N - 1;             // clamp: keeps ASYNCcnt uniform
    const __bf16* src = Bw + (long)nrow * K + kc + cl;
#if HAVE_ASYNC_LDS
    async_b128(src, &sB[r][cl]);
#else
    *(uint4*)&sB[r][cl] = *(const uint4*)src;
#endif
  }
}

__global__ __launch_bounds__(256) void gemm_bf16_kernel(
    const __bf16* __restrict__ A, const __bf16* __restrict__ Bw,
    const float* __restrict__ bias, __bf16* __restrict__ Cb,
    float* __restrict__ Cf, int M, int N, int K) {
  __shared__ __align__(16) __bf16 sA[2][G_BM][G_LDT];
  __shared__ __align__(16) __bf16 sB[2][G_BN][G_LDT];

  const int tiles_n = (N + G_BN - 1) / G_BN;
  const int bm = blockIdx.x / tiles_n;
  const int bn = blockIdx.x % tiles_n;
  const int m_base = bm * G_BM, n_base = bn * G_BN;

  const int tid = threadIdx.x, lane = tid & 31, wave = tid >> 5;
  const int wm = wave >> 1, wn = wave & 1;   // 4x2 wave grid, 32x32 per wave

  v8f acc[2][2];
  acc[0][0] = {}; acc[0][1] = {}; acc[1][0] = {}; acc[1][1] = {};

  const int nc = K >> 5;

#if HAVE_ASYNC_LDS
  stage_tile(A, Bw, sA[0], sB[0], m_base, n_base, N, K, 0, tid);
  if (nc > 1) stage_tile(A, Bw, sA[1], sB[1], m_base, n_base, N, K, 1, tid);
  for (int c = 0; c < nc; ++c) {
    if (c + 1 < nc) WAIT_ASYNC(3); else WAIT_ASYNC(0);
    __syncthreads();
    const __bf16* pa = &sA[c & 1][0][0];
    const __bf16* pb = &sB[c & 1][0][0];
    v16bf b0 = frag_b_lds(pb, G_LDT, wn * 32,      0, lane);
    v16bf b1 = frag_b_lds(pb, G_LDT, wn * 32 + 16, 0, lane);
    v16bf a0 = frag_a_lds(pa, G_LDT, wm * 32,      0, lane);
    v16bf a1 = frag_a_lds(pa, G_LDT, wm * 32 + 16, 0, lane);
    acc[0][0] = wmma_bf16(a0, b0, acc[0][0]);
    acc[0][1] = wmma_bf16(a0, b1, acc[0][1]);
    acc[1][0] = wmma_bf16(a1, b0, acc[1][0]);
    acc[1][1] = wmma_bf16(a1, b1, acc[1][1]);
    __syncthreads();
    if (c + 2 < nc)
      stage_tile(A, Bw, sA[c & 1], sB[c & 1], m_base, n_base, N, K, c + 2, tid);
  }
#else
  for (int c = 0; c < nc; ++c) {
    stage_tile(A, Bw, sA[0], sB[0], m_base, n_base, N, K, c, tid);
    __syncthreads();
    const __bf16* pa = &sA[0][0][0];
    const __bf16* pb = &sB[0][0][0];
    v16bf b0 = frag_b_lds(pb, G_LDT, wn * 32,      0, lane);
    v16bf b1 = frag_b_lds(pb, G_LDT, wn * 32 + 16, 0, lane);
    v16bf a0 = frag_a_lds(pa, G_LDT, wm * 32,      0, lane);
    v16bf a1 = frag_a_lds(pa, G_LDT, wm * 32 + 16, 0, lane);
    acc[0][0] = wmma_bf16(a0, b0, acc[0][0]);
    acc[0][1] = wmma_bf16(a0, b1, acc[0][1]);
    acc[1][0] = wmma_bf16(a1, b0, acc[1][0]);
    acc[1][1] = wmma_bf16(a1, b1, acc[1][1]);
    __syncthreads();
  }
#endif

#pragma unroll
  for (int nt = 0; nt < 2; ++nt) {
    const int col = n_base + wn * 32 + nt * 16 + (lane & 15);
    if (col < N) {
      const float bv = bias ? bias[col] : 0.f;
#pragma unroll
      for (int mt = 0; mt < 2; ++mt) {
#pragma unroll
        for (int r = 0; r < 8; ++r) {
          int row = m_base + wm * 32 + mt * 16 + r + ((lane >> 4) << 3);
          long idx = (long)row * N + col;
          float v = acc[mt][nt][r] + bv;
          if (Cf) Cf[idx] = v;
          if (Cb) Cb[idx] = (__bf16)v;
        }
      }
    }
  }
}

// ---------------------------------------------------------------------------
// Persistent BiLSTM recurrence kernel.
// Grid = 32 WGs: dir = blk>>4 (0=fwd,1=bwd), j = blk&15 (32-wide H column tile).
// Each wave pins its Wh slice in VGPRs as 16 WMMA B-fragments (reused 512x).
// Per-direction step ordering via device-scope atomic split barrier (16 WGs),
// h double-buffered in global (bf16).
// ---------------------------------------------------------------------------
__device__ __forceinline__ void dir_grid_sync(unsigned* cnt, unsigned* gen,
                                              unsigned nwg) {
  __threadfence();
  __syncthreads();
  if (threadIdx.x == 0) {
    unsigned g = __atomic_load_n(gen, __ATOMIC_ACQUIRE);
    if (atomicAdd(cnt, 1u) == nwg - 1u) {
      __atomic_store_n(cnt, 0u, __ATOMIC_RELAXED);
      __atomic_store_n(gen, g + 1u, __ATOMIC_RELEASE);
    } else {
      while (__atomic_load_n(gen, __ATOMIC_ACQUIRE) == g)
        __builtin_amdgcn_s_sleep(1);
    }
  }
  __syncthreads();
}

#define H_LDT 528   // 512 + 16

__global__ __launch_bounds__(256) void bilstm_kernel(
    const __bf16* __restrict__ xpF, const __bf16* __restrict__ xpB, // [S*B,2048]
    const __bf16* __restrict__ WhF, const __bf16* __restrict__ WhB, // [2048,512]
    __bf16* __restrict__ hbuf,      // [2 dir][2 phase][32*512] bf16 (pre-zeroed)
    __bf16* __restrict__ lstm_o,    // [B*S, 1024] bf16
    unsigned* __restrict__ bar_cnt, unsigned* __restrict__ bar_gen) {
  const int dir = blockIdx.x >> 4;
  const int j   = blockIdx.x & 15;
  const __bf16* xp = dir ? xpB : xpF;
  const __bf16* Wh = dir ? WhB : WhF;
  __bf16* hb = hbuf + (long)dir * 2 * NER_B * NER_H;
  unsigned* cnt = bar_cnt + dir;
  unsigned* gen = bar_gen + dir;

  __shared__ __align__(16) __bf16 hL[32][H_LDT];
  __shared__ float zL[32][132];
  __shared__ float cL[32][32];

  const int tid = threadIdx.x, lane = tid & 31, wave = tid >> 5;

  for (int i = tid; i < 32 * 32; i += 256) cL[i >> 5][i & 31] = 0.f;

  // Pin Wh B-fragments in registers: wave covers z-cols n = 16*wave .. +15
  const int nloc = 16 * wave + (lane & 15);
  const int g_gate = nloc >> 5;
  const int hcol   = nloc & 31;
  const long wrow  = (long)g_gate * NER_H + j * 32 + hcol;  // row of flat Wh
  v16bf bw[16];
#pragma unroll
  for (int kc = 0; kc < 16; ++kc) {
    const __bf16* pw = Wh + wrow * NER_H + kc * 32 + kb_off(lane);
#pragma unroll
    for (int e = 0; e < 16; ++e) bw[kc][e] = pw[e];
  }

  for (int t = 0; t < NER_S; ++t) {
    const int ph = t & 1;
    // full h_prev [32,512] -> LDS
    const __bf16* hsrc = hb + (long)ph * NER_B * NER_H;
#if HAVE_ASYNC_LDS
    for (int v = tid; v < 2048; v += 256) {
      int i = v * 8, r = i >> 9, c = i & 511;
      async_b128(hsrc + i, &hL[r][c]);
    }
    WAIT_ASYNC(0);
#else
    for (int v = tid; v < 2048; v += 256) {
      int i = v * 8, r = i >> 9, c = i & 511;
      *(uint4*)&hL[r][c] = *(const uint4*)(hsrc + i);
    }
#endif
    __syncthreads();

    v8f acc0 = {}, acc1 = {};
#pragma unroll
    for (int kc = 0; kc < 16; ++kc) {
      v16bf a0 = frag_a_lds(&hL[0][0], H_LDT, 0,  kc * 32, lane);
      v16bf a1 = frag_a_lds(&hL[0][0], H_LDT, 16, kc * 32, lane);
      acc0 = wmma_bf16(a0, bw[kc], acc0);
      acc1 = wmma_bf16(a1, bw[kc], acc1);
    }
    {
      const int zc = 16 * wave + (lane & 15);
#pragma unroll
      for (int r = 0; r < 8; ++r) {
        int rr = r + ((lane >> 4) << 3);
        zL[rr][zc]      = acc0[r];
        zL[16 + rr][zc] = acc1[r];
      }
    }
    __syncthreads();

    const int s = dir ? (NER_S - 1 - t) : t;
    for (int i2 = tid; i2 < 1024; i2 += 256) {
      const int b = i2 >> 5, c2 = i2 & 31;
      const long xr = ((long)s * NER_B + b) * NER_G + j * 32 + c2;
      float zf = zL[b][c2]      + (float)xp[xr];
      float zi = zL[b][32 + c2] + (float)xp[xr + 512];
      float zo = zL[b][64 + c2] + (float)xp[xr + 1024];
      float zg = zL[b][96 + c2] + (float)xp[xr + 1536];
      float f  = 1.f / (1.f + __expf(-zf));
      float ii = 1.f / (1.f + __expf(-zi));
      float oo = 1.f / (1.f + __expf(-zo));
      float gg = tanhf(zg);
      float cv = f * cL[b][c2] + ii * gg;
      cL[b][c2] = cv;
      __bf16 hv = (__bf16)(oo * tanhf(cv));
      hb[(long)(1 - ph) * NER_B * NER_H + b * NER_H + j * 32 + c2] = hv;
      lstm_o[((long)b * NER_S + s) * (2 * NER_H) + dir * NER_H + j * 32 + c2] = hv;
    }
    dir_grid_sync(cnt, gen, 16u);
  }
}

// ---------------------------------------------------------------------------
// Fused attention: per (head, batch, 16 query rows):
//   scores = (Q Kc^T)/sqrt(D) -> softmax -> O = P V    (all WMMA bf16)
// ---------------------------------------------------------------------------
__global__ __launch_bounds__(256) void attn_kernel(
    const __bf16* __restrict__ Q, const __bf16* __restrict__ Kc,
    const __bf16* __restrict__ V, __bf16* __restrict__ O) {
  const int bid = blockIdx.x;
  const int rt = bid & 31;             // 32 row-tiles of 16
  const int b  = (bid >> 5) & 31;
  const int h  = bid >> 10;
  const int r0 = rt * 16;

  __shared__ __align__(16) float  sc[16][516];
  __shared__ __align__(16) __bf16 pr[16][520];
  __shared__ float red[16][16];

  const int tid = threadIdx.x, lane = tid & 31, wave = tid >> 5;

  // ---- scores: M=16, N=512 (wave handles 4 n-tiles), K=288 ----
  v8f acc[4]; acc[0] = {}; acc[1] = {}; acc[2] = {}; acc[3] = {};
  const long qrow = ((long)b * NER_S + r0 + (lane & 15)) * NER_F + h * NER_D;
#pragma unroll 1
  for (int kc = 0; kc < 9; ++kc) {
    v16bf a;
#pragma unroll
    for (int e = 0; e < 16; ++e) a[e] = Q[qrow + kc * 32 + ka_off(e, lane)];
#pragma unroll
    for (int nt = 0; nt < 4; ++nt) {
      const int key = (wave * 4 + nt) * 16 + (lane & 15);
      const __bf16* kp = Kc + ((long)b * NER_S + key) * NER_F + h * NER_D +
                         kc * 32 + kb_off(lane);
      v16bf bb;
#pragma unroll
      for (int e = 0; e < 16; ++e) bb[e] = kp[e];
      acc[nt] = wmma_bf16(a, bb, acc[nt]);
    }
  }
  const float scale = 0.0589255650989f;  // 1/sqrt(288)
#pragma unroll
  for (int nt = 0; nt < 4; ++nt) {
    const int ccol = (wave * 4 + nt) * 16 + (lane & 15);
#pragma unroll
    for (int r = 0; r < 8; ++r)
      sc[r + ((lane >> 4) << 3)][ccol] = acc[nt][r] * scale;
  }
  __syncthreads();

  // ---- softmax over 512 cols per row (16 threads/row, 32 cols each) ----
  const int row = tid >> 4, sub = tid & 15;
  float mx = -1e30f;
  for (int c = sub * 32; c < sub * 32 + 32; ++c) mx = fmaxf(mx, sc[row][c]);
  red[row][sub] = mx;
  __syncthreads();
  if (sub == 0) {
    float m2 = red[row][0];
#pragma unroll
    for (int q = 1; q < 16; ++q) m2 = fmaxf(m2, red[row][q]);
    red[row][0] = m2;
  }
  __syncthreads();
  const float rmax = red[row][0];
  float sum = 0.f;
  for (int c = sub * 32; c < sub * 32 + 32; ++c) {
    float e = __expf(sc[row][c] - rmax);
    sc[row][c] = e;
    sum += e;
  }
  __syncthreads();
  red[row][sub] = sum;
  __syncthreads();
  if (sub == 0) {
    float s2 = 0.f;
#pragma unroll
    for (int q = 0; q < 16; ++q) s2 += red[row][q];
    red[row][0] = s2;
  }
  __syncthreads();
  const float inv = 1.f / red[row][0];
  for (int c = sub * 32; c < sub * 32 + 32; ++c)
    pr[row][c] = (__bf16)(sc[row][c] * inv);
  __syncthreads();

  // ---- O = P V : M=16, N=288 (18 n-tiles round-robin), K=512 ----
  for (int nt = wave; nt < 18; nt += 8) {
    const int n0 = nt * 16;
    v8f oacc = {};
#pragma unroll 1
    for (int kc = 0; kc < 16; ++kc) {
      v16bf a = frag_a_lds(&pr[0][0], 520, 0, kc * 32, lane);
      const __bf16* vp = V + ((long)b * NER_S + kc * 32 + kb_off(lane)) * NER_F +
                         h * NER_D + n0 + (lane & 15);
      v16bf bb;
#pragma unroll
      for (int e = 0; e < 16; ++e) bb[e] = vp[(long)e * NER_F];
      oacc = wmma_bf16(a, bb, oacc);
    }
    const int cc = h * NER_D + n0 + (lane & 15);
#pragma unroll
    for (int r = 0; r < 8; ++r) {
      int rr = r0 + r + ((lane >> 4) << 3);
      O[((long)b * NER_S + rr) * NER_F + cc] = (__bf16)oacc[r];
    }
  }
}

// ---------------------------------------------------------------------------
// Host orchestration
// ---------------------------------------------------------------------------
extern "C" void kernel_launch(void* const* d_in, const int* in_sizes, int n_in,
                              void* d_out, int out_size, void* d_ws,
                              size_t ws_size, hipStream_t stream) {
  const float* x    = (const float*)d_in[0];
  const float* Wx_f = (const float*)d_in[1];
  const float* Wh_f = (const float*)d_in[2];
  const float* b_f  = (const float*)d_in[3];
  const float* Wx_b = (const float*)d_in[4];
  const float* Wh_b = (const float*)d_in[5];
  const float* b_b  = (const float*)d_in[6];
  const float* We   = (const float*)d_in[7];
  const float* be   = (const float*)d_in[8];
  const float* Wq   = (const float*)d_in[9];
  const float* bq   = (const float*)d_in[10];
  const float* Wk   = (const float*)d_in[11];
  const float* bk   = (const float*)d_in[12];
  const float* Wv   = (const float*)d_in[13];
  const float* bv   = (const float*)d_in[14];
  const float* Wc   = (const float*)d_in[15];
  const float* bc   = (const float*)d_in[16];

  char* base = (char*)d_ws;
  size_t off = 0;
  auto alloc = [&](size_t bytes) -> void* {
    void* r = base + off;
    off = (off + bytes + 255) & ~(size_t)255;
    return r;
  };
  const long MS = (long)NER_B * NER_S;  // 16384
  __bf16* xT   = (__bf16*)alloc(MS * NER_E * 2);
  __bf16* WxFb = (__bf16*)alloc((long)NER_G * NER_E * 2);
  __bf16* WxBb = (__bf16*)alloc((long)NER_G * NER_E * 2);
  __bf16* WhFb = (__bf16*)alloc((long)NER_G * NER_H * 2);
  __bf16* WhBb = (__bf16*)alloc((long)NER_G * NER_H * 2);
  __bf16* Web  = (__bf16*)alloc((long)NER_D * 2 * NER_H * 2);
  __bf16* Wqb  = (__bf16*)alloc((long)NER_F * 2 * NER_H * 2);
  __bf16* Wkb  = (__bf16*)alloc((long)NER_F * NER_D * 2);
  __bf16* Wvb  = (__bf16*)alloc((long)NER_F * NER_D * 2);
  __bf16* Wcb  = (__bf16*)alloc((long)9 * NER_F * 2);
  __bf16* xpF  = (__bf16*)alloc(MS * NER_G * 2);
  __bf16* xpB  = (__bf16*)alloc(MS * NER_G * 2);
  __bf16* hbuf = (__bf16*)alloc((long)2 * 2 * NER_B * NER_H * 2);
  __bf16* lstm = (__bf16*)alloc(MS * 2 * NER_H * 2);
  __bf16* ec   = (__bf16*)alloc(MS * NER_D * 2);
  __bf16* Qb   = (__bf16*)alloc(MS * NER_F * 2);
  __bf16* Kb   = (__bf16*)alloc(MS * NER_F * 2);
  __bf16* Vb   = (__bf16*)alloc(MS * NER_F * 2);
  __bf16* Ob   = (__bf16*)alloc(MS * NER_F * 2);
  unsigned* bar_cnt = (unsigned*)alloc(8);
  unsigned* bar_gen = (unsigned*)alloc(8);

  // barrier + h0 init (bf16 zero == 0x0000)
  (void)hipMemsetAsync(bar_cnt, 0, 8, stream);
  (void)hipMemsetAsync(bar_gen, 0, 8, stream);
  (void)hipMemsetAsync(hbuf, 0, (size_t)2 * 2 * NER_B * NER_H * 2, stream);

  auto cvt = [&](const float* in, __bf16* out, long n) {
    cvt_bf16_kernel<<<dim3(512), dim3(256), 0, stream>>>(in, out, n);
  };
  cvt(Wx_f, WxFb, (long)NER_G * NER_E);
  cvt(Wx_b, WxBb, (long)NER_G * NER_E);
  cvt(Wh_f, WhFb, (long)NER_G * NER_H);
  cvt(Wh_b, WhBb, (long)NER_G * NER_H);
  cvt(We,   Web,  (long)NER_D * 2 * NER_H);
  cvt(Wq,   Wqb,  (long)NER_F * 2 * NER_H);
  cvt(Wk,   Wkb,  (long)NER_F * NER_D);
  cvt(Wv,   Wvb,  (long)NER_F * NER_D);
  cvt(Wc,   Wcb,  (long)9 * NER_F);

  xpose_x_kernel<<<dim3(NER_B * NER_S), dim3(256), 0, stream>>>(x, xT);

  auto gemm = [&](const __bf16* A, const __bf16* Bw, const float* bias,
                  __bf16* Cb, float* Cf, int M, int N, int K) {
    int tn = (N + G_BN - 1) / G_BN, tm = M / G_BM;
    gemm_bf16_kernel<<<dim3(tm * tn), dim3(256), 0, stream>>>(A, Bw, bias, Cb,
                                                              Cf, M, N, K);
  };

  // input projections for both LSTM directions: [S*B, 4H]
  gemm(xT, WxFb, b_f, xpF, nullptr, (int)MS, NER_G, NER_E);
  gemm(xT, WxBb, b_b, xpB, nullptr, (int)MS, NER_G, NER_E);

  // persistent BiLSTM recurrence (16 WGs per direction + split barrier)
  bilstm_kernel<<<dim3(32), dim3(256), 0, stream>>>(xpF, xpB, WhFb, WhBb, hbuf,
                                                    lstm, bar_cnt, bar_gen);

  // experts, then attention projections
  gemm(lstm, Web, be, ec, nullptr, (int)MS, NER_D, 2 * NER_H);
  gemm(lstm, Wqb, bq, Qb, nullptr, (int)MS, NER_F, 2 * NER_H);
  gemm(ec,   Wkb, bk, Kb, nullptr, (int)MS, NER_F, NER_D);
  gemm(ec,   Wvb, bv, Vb, nullptr, (int)MS, NER_F, NER_D);

  // fused multi-head attention
  attn_kernel<<<dim3(4 * NER_B * (NER_S / 16)), dim3(256), 0, stream>>>(Qb, Kb,
                                                                        Vb, Ob);

  // final classifier -> f32 logits [B,S,9]
  gemm(Ob, Wcb, bc, nullptr, (float*)d_out, (int)MS, 9, NER_F);

  (void)in_sizes; (void)n_in; (void)out_size; (void)ws_size;
}
